// MetropolisHastingsPretrain_54966991454580
// MI455X (gfx1250) — compile-verified, matrix operand order checked by reading, not run on hardware
//
#include <hip/hip_runtime.h>
#include <hip/hip_bf16.h>

// ---------------------------------------------------------------------------
// MetropolisHastingsPretrain on MI455X (gfx1250, wave32, WMMA)
//
// Roofline: 11 passes x 65536 x (2*48*256) ~= 17.7 GFLOP matmul vs ~25 MB HBM
// (12.6 MB sample read + 12.6 MB write; weights L2-resident). ~700 FLOP/B ->
// compute bound on the matrix pipe. Strategy: fp16 WMMA w/ f32 accumulate,
// weights + walker state pinned in LDS for the entire 10-step chain.
// This round: branch-free tanh (native v_tanh_f32 if available) to strip the
// libm EXEC-divergent expansion out of the WMMA inner loop.
// ---------------------------------------------------------------------------

typedef __attribute__((ext_vector_type(16))) _Float16 v16h;
typedef __attribute__((ext_vector_type(8)))  float    v8f;

union AFrag {
    v16h  v;
    float4 f4[2];   // two 16B LDS loads fill the 8-VGPR fragment
};

#define TPB   256            // 8 wave32s per block
#define SPB   128            // samples per block (8 M-tiles of 16)
#define DPAD  64             // K = 48 padded to 64 (two 16x16x32 k-chunks)
#define HDIM  256

// ---- branch-free tanh: native trans op on gfx1250, exp-based fallback -----
__device__ __forceinline__ float fast_tanh(float x) {
#if __has_builtin(__builtin_amdgcn_tanhf)
    return __builtin_amdgcn_tanhf(x);          // v_tanh_f32 (TRANS, co-issues)
#else
    // 1 - 2/(exp(2x)+1): v_exp_f32 + v_rcp path, no EXEC divergence.
    // exp(+inf)->inf -> 1 ; exp(-inf)->0 -> -1 ; monotone in between.
    return 1.0f - 2.0f / (__expf(2.0f * x) + 1.0f);
#endif
}

// ---- counter-based RNG (jax.random not reproducible; structure-equivalent) --
__device__ __forceinline__ unsigned pcg_hash(unsigned x) {
    x = x * 747796405u + 2891336453u;
    x = ((x >> ((x >> 28) + 4u)) ^ x) * 277803737u;
    return (x >> 22) ^ x;
}
__device__ __forceinline__ float u01(unsigned h) {
    return (float)(h >> 8) * (1.0f / 16777216.0f) + (0.5f / 16777216.0f);
}
__device__ __forceinline__ float rng_normal(unsigned id, unsigned step) {
    unsigned h1 = pcg_hash(id ^ (step * 0x9E3779B9u) ^ 0x85EBCA6Bu);
    unsigned h2 = pcg_hash(h1 + id + 0xC2B2AE35u);
    float r = __builtin_sqrtf(fmaxf(-2.0f * __logf(u01(h1)), 0.0f));
    return r * __cosf(6.28318530718f * u01(h2));
}

// ---------------------------------------------------------------------------
// One full-block pass of  p = exp(2 * tanh(A @ W + b) . w)  for 128 rows.
// sA: [128][64] f16 (K-padded activations)   sW: [256][64] f16 (W^T, padded)
// Each wave handles one 16-row M-tile, sweeps 16 N-tiles x 2 K-chunks.
// EXEC is all-ones at every WMMA (no divergence on this path).
// ---------------------------------------------------------------------------
__device__ __forceinline__ void block_logamp_p(
    const _Float16* __restrict__ sA,
    const _Float16* __restrict__ sW,
    const float*    __restrict__ sB,
    const float*    __restrict__ sw2,
    float*          __restrict__ pOut,
    int lane, int wave)
{
    const int hs  = lane >> 4;     // half-wave select (K sub-pattern)
    const int lm  = lane & 15;     // row (A) / col (B,C)
    const int row = wave * 16 + lm;

    // A fragments: K-chunk 0 (K=0..31) and chunk 1 (K=32..63, zero-padded tail)
    AFrag a0, a1;
    a0.f4[0] = *(const float4*)(sA + row * DPAD +      hs * 8);
    a0.f4[1] = *(const float4*)(sA + row * DPAD + 16 + hs * 8);
    a1.f4[0] = *(const float4*)(sA + row * DPAD + 32 + hs * 8);
    a1.f4[1] = *(const float4*)(sA + row * DPAD + 48 + hs * 8);

    float part[8];
#pragma unroll
    for (int v = 0; v < 8; ++v) part[v] = 0.0f;

#pragma unroll
    for (int t = 0; t < 16; ++t) {
        const int n = t * 16 + lm;                 // this lane's output column
        AFrag b0, b1;
        b0.f4[0] = *(const float4*)(sW + n * DPAD +      hs * 8);
        b0.f4[1] = *(const float4*)(sW + n * DPAD + 16 + hs * 8);
        b1.f4[0] = *(const float4*)(sW + n * DPAD + 32 + hs * 8);
        b1.f4[1] = *(const float4*)(sW + n * DPAD + 48 + hs * 8);

        v8f c = {};
        c = __builtin_amdgcn_wmma_f32_16x16x32_f16(false, a0.v, false, b0.v,
                                                   (short)0, c, false, false);
        c = __builtin_amdgcn_wmma_f32_16x16x32_f16(false, a1.v, false, b1.v,
                                                   (short)0, c, false, false);

        const float wn = sw2[n];
        const float bn = sB[n];
#pragma unroll
        for (int v = 0; v < 8; ++v)
            part[v] += fast_tanh(c[v] + bn) * wn;  // TRANS VALU co-issues w/ XDL
    }

    // reduce the 16 columns each lane covers across its 16-lane half
#pragma unroll
    for (int v = 0; v < 8; ++v) {
        part[v] += __shfl_xor(part[v], 1, 32);
        part[v] += __shfl_xor(part[v], 2, 32);
        part[v] += __shfl_xor(part[v], 4, 32);
        part[v] += __shfl_xor(part[v], 8, 32);
    }
    if (lm == 0) {                                 // lanes 0 and 16
#pragma unroll
        for (int v = 0; v < 8; ++v)
            pOut[wave * 16 + hs * 8 + v] = __expf(2.0f * part[v]);
    }
}

__global__ __launch_bounds__(TPB) void mh_pretrain_kernel(
    const float* __restrict__ cs,                  // [N][48]
    const float* __restrict__ W1, const float* __restrict__ b1,
    const float* __restrict__ w2,
    const float* __restrict__ Wh, const float* __restrict__ bh,
    const float* __restrict__ wh2,
    float* __restrict__ out, int n)
{
    __shared__ _Float16 sW[HDIM * DPAD];           // 32 KB  W^T f16, K-padded
    __shared__ _Float16 sA[SPB  * DPAD];           // 16 KB  activation staging
    __shared__ _Float16 sS[SPB  * 48];             // 12 KB  chain state
    __shared__ float    sB[HDIM], sw2[HDIM];       //  2 KB
    __shared__ float    pCur[SPB], pProp[SPB];     //  1 KB
    __shared__ unsigned accF[SPB];                 //  0.5 KB   (total 63.5 KB)

    const int tid  = threadIdx.x;
    const int lane = tid & 31;
    const int wave = tid >> 5;
    const int blk  = blockIdx.x;
    const bool firstHalf = blk < (int)(gridDim.x >> 1);

    const float* W  = firstHalf ? W1 : Wh;
    const float* bb = firstHalf ? b1 : bh;
    const float* wv = firstHalf ? w2 : wh2;
    const long  base = (long)blk * SPB;            // global sample base
    const unsigned pmask = (unsigned)(n - 1);      // n = 2^16 -> perm mask

    // hint the streaming sample tile into cache ahead of the staging loop
    __builtin_prefetch(&cs[base * 48 + tid * 24], 0, 1);   // global_prefetch_b8

    // --- stage weights: W (48x256 row-major) -> W^T f16 [256][64], pad K ---
    {
        const int col = tid;                       // one H-column per thread
        for (int k = 0; k < 48; ++k)
            sW[col * DPAD + k] = (_Float16)W[k * HDIM + col];
#pragma unroll
        for (int k = 48; k < DPAD; ++k)
            sW[col * DPAD + k] = (_Float16)0.0f;
        sB[col]  = bb[col];
        sw2[col] = wv[col];
    }
    // --- stage walkers (f32 -> f16) into state + activation buffers ---
    for (int e = tid; e < SPB * 48; e += TPB) {
        const int s = e / 48, d = e % 48;
        const float x = cs[(base + s) * 48 + d];
        sS[s * 48 + d]   = (_Float16)x;
        sA[s * DPAD + d] = (_Float16)x;
    }
    for (int e = tid; e < SPB * 16; e += TPB) {    // zero K-pad 48..63
        const int s = e / 16, d = 48 + (e % 16);
        sA[s * DPAD + d] = (_Float16)0.0f;
    }
    __syncthreads();

    // --- initial amplitudes ---
    block_logamp_p(sA, sW, sB, sw2, pCur, lane, wave);
    __syncthreads();

    // --- 10-step MH chain, state never leaves LDS ---
    for (int step = 0; step < 10; ++step) {
        // propose: new = s + sigma * z
        for (int e = tid; e < SPB * 48; e += TPB) {
            const int s = e / 48, d = e % 48;
            const unsigned gid = (unsigned)((base + s) * 48 + d);
            const float z = rng_normal(gid, (unsigned)step);
            const float x = (float)sS[s * 48 + d] + 0.02f * z;
            sA[s * DPAD + d] = (_Float16)x;
        }
        __syncthreads();

        block_logamp_p(sA, sW, sB, sw2, pProp, lane, wave);
        __syncthreads();

        // accept/reject
        if (tid < SPB) {
            const unsigned gid = (unsigned)(base + tid);
            const float u = u01(pcg_hash(gid ^ ((step + 17u) * 0x632BE5ABu)));
            const unsigned a = (pProp[tid] > u * pCur[tid]) ? 1u : 0u;
            accF[tid] = a;
            if (a) pCur[tid] = pProp[tid];
        }
        __syncthreads();

        // commit accepted proposals into chain state
        for (int e = tid; e < SPB * 48; e += TPB) {
            const int s = e / 48, d = e % 48;
            if (accF[s]) sS[s * 48 + d] = sA[s * DPAD + d];
        }
        __syncthreads();
    }

    // --- scatter through a bijective pseudo-permutation of [0, 2^16) ---
    for (int e = tid; e < SPB * 48; e += TPB) {
        const int s = e / 48, d = e % 48;
        const unsigned gi = (unsigned)(base + s);
        const unsigned pi = (gi * 40503u + 2077u) & pmask;  // odd mult -> bijection
        out[(size_t)pi * 48 + d] = (float)sS[s * 48 + d];
    }
}

extern "C" void kernel_launch(void* const* d_in, const int* in_sizes, int n_in,
                              void* d_out, int out_size, void* d_ws, size_t ws_size,
                              hipStream_t stream) {
    const float* cs  = (const float*)d_in[0];   // curr_sample (N,16,3) f32
    const float* W1  = (const float*)d_in[1];   // (48,256)
    const float* b1  = (const float*)d_in[2];   // (256,)
    const float* w2  = (const float*)d_in[3];   // (256,)
    const float* Wh  = (const float*)d_in[4];   // (48,256)
    const float* bh  = (const float*)d_in[5];   // (256,)
    const float* wh2 = (const float*)d_in[6];   // (256,)
    float* out = (float*)d_out;

    const int n = in_sizes[0] / 48;             // 65536 walkers
    const int blocks = n / SPB;                 // 512 blocks; first half m, second h
    mh_pretrain_kernel<<<blocks, TPB, 0, stream>>>(cs, W1, b1, w2, Wh, bh, wh2,
                                                   out, n);
}
